// CausalSelfAttention_5420248728252
// MI455X (gfx1250) — matrix operand order repaired and checked by reading, not compiled, not run on hardware
//
#include <hip/hip_runtime.h>
#include <hip/hip_bf16.h>

// Problem constants (reference: B=2, T=2048, C=1024, H=16, D=64)
#define BATCH  2
#define T_SEQ  2048
#define C_DIM  1024
#define H_NUM  16
#define D_HEAD 64
#define M_TOK  (BATCH * T_SEQ)   // 4096
#define N_QKV  (3 * C_DIM)       // 3072

typedef __attribute__((ext_vector_type(16))) _Float16 v16h;
typedef __attribute__((ext_vector_type(8)))  float    v8f;

#define WMMA16(A, B, C) \
  __builtin_amdgcn_wmma_f32_16x16x32_f16(false, (A), false, (B), (short)0, (C), false, false)

union FragU { v16h v; uint4 u[2]; };

// Load a 16-half fragment as two 16B chunks (elements 0..7 from p0, 8..15 from p1).
__device__ __forceinline__ v16h ld_frag(const _Float16* p0, const _Float16* p1) {
  FragU f;
  f.u[0] = *reinterpret_cast<const uint4*>(p0);
  f.u[1] = *reinterpret_cast<const uint4*>(p1);
  return f.v;
}

// ---------------------------------------------------------------------------
// Kernel 1a: f32 -> f16 copy
__global__ void cvt_f16_kernel(const float* __restrict__ src,
                               _Float16* __restrict__ dst, int n) {
  int i = blockIdx.x * blockDim.x + threadIdx.x;
  if (i < n) dst[i] = (_Float16)src[i];
}

// Kernel 1b: transpose [rows,cols] f32 -> [cols,rows] f16 (read-coalesced)
__global__ void transpose_f16_kernel(const float* __restrict__ src,
                                     _Float16* __restrict__ dst,
                                     int rows, int cols) {
  int i = blockIdx.x * blockDim.x + threadIdx.x;
  if (i < rows * cols) {
    int r = i / cols, c = i - r * cols;
    dst[(size_t)c * rows + r] = (_Float16)src[i];
  }
}

// ---------------------------------------------------------------------------
// Kernel 2: QKV GEMM. A=[M_TOK,C] f16, Bt=[N_QKV,C] f16 (W^T). Each wave owns
// a 32x64 tile (2 M-tiles x 4 N-tiles of 16x16). 4 waves/block -> 128x64 block.
// Epilogue scatters into q[B,H,T,D] (scaled 1/sqrt(D)), k[B,H,T,D], vT[B,H,D,T].
__global__ void gemm_qkv_kernel(const _Float16* __restrict__ A,
                                const _Float16* __restrict__ Bt,
                                const float* __restrict__ bias,
                                _Float16* __restrict__ qh,
                                _Float16* __restrict__ kh,
                                _Float16* __restrict__ vT) {
  const int lane = threadIdx.x & 31;
  const int wv   = threadIdx.x >> 5;
  const int hl   = lane >> 4;     // lane half (0/1)
  const int l15  = lane & 15;
  const int n0   = blockIdx.x * 64;
  const int m0   = blockIdx.y * 128 + wv * 32;

  const _Float16* arow0 = A + (size_t)(m0 + l15) * C_DIM;
  const _Float16* arow1 = A + (size_t)(m0 + 16 + l15) * C_DIM;
  const _Float16* bcol[4];
#pragma unroll
  for (int nt = 0; nt < 4; ++nt)
    bcol[nt] = Bt + (size_t)(n0 + nt * 16 + l15) * C_DIM;

  v8f z = {};
  v8f acc[2][4];
#pragma unroll
  for (int i = 0; i < 2; ++i)
#pragma unroll
    for (int nt = 0; nt < 4; ++nt) acc[i][nt] = z;

  for (int kb = 0; kb < C_DIM; kb += 32) {
    __builtin_prefetch(arow0 + kb + 512, 0, 1);
    __builtin_prefetch(bcol[0] + kb + 512, 0, 1);
    // A fragment: VGPR j<4 -> K = 8*hl + 2j..; VGPR j>=4 -> K = 16 + 8*hl + ..
    v16h a0 = ld_frag(arow0 + kb + 8 * hl, arow0 + kb + 16 + 8 * hl);
    v16h a1 = ld_frag(arow1 + kb + 8 * hl, arow1 + kb + 16 + 8 * hl);
#pragma unroll
    for (int nt = 0; nt < 4; ++nt) {
      // B fragment: K = 8*hl.. and 16 + 8*hl.. (W stored transposed)
      const _Float16* bp = bcol[nt] + kb + 8 * hl;
      v16h b = ld_frag(bp, bp + 16);
      acc[0][nt] = WMMA16(a0, b, acc[0][nt]);
      acc[1][nt] = WMMA16(a1, b, acc[1][nt]);
    }
  }

  const int j = n0 >> 10;                  // 0=q, 1=k, 2=v (64 | 1024)
  const int h = (n0 & (C_DIM - 1)) >> 6;   // head index
  float bs[4];
#pragma unroll
  for (int nt = 0; nt < 4; ++nt) bs[nt] = bias[n0 + nt * 16 + l15];

#pragma unroll
  for (int i = 0; i < 2; ++i)
#pragma unroll
    for (int nt = 0; nt < 4; ++nt)
#pragma unroll
      for (int r = 0; r < 8; ++r) {
        int row = m0 + i * 16 + r + 8 * hl;   // C layout: M = r + 8*half
        int b   = row >> 11;                  // token -> batch
        int t   = row & (T_SEQ - 1);
        int d   = nt * 16 + l15;              // N = lane&15
        float v = acc[i][nt][r] + bs[nt];
        size_t bh = (size_t)(b * H_NUM + h);
        if (j == 0)
          qh[(bh * T_SEQ + t) * D_HEAD + d] = (_Float16)(v * 0.125f);
        else if (j == 1)
          kh[(bh * T_SEQ + t) * D_HEAD + d] = (_Float16)v;
        else
          vT[(bh * D_HEAD + d) * T_SEQ + t] = (_Float16)v;
      }
}

// ---------------------------------------------------------------------------
// Kernel 3: flash attention. One wave per 16-query chunk, 8 waves/block.
// Streams key blocks of 32: S (16x32) via 4 WMMAs, online softmax with
// shfl_xor row reductions, P relaid out via per-wave LDS, P.V via 4 WMMAs.
__global__ void flash_attn_kernel(const _Float16* __restrict__ qh,
                                  const _Float16* __restrict__ kh,
                                  const _Float16* __restrict__ vT,
                                  _Float16* __restrict__ yh) {
  const int lane = threadIdx.x & 31;
  const int wv   = threadIdx.x >> 5;
  const int hl   = lane >> 4;
  const int l15  = lane & 15;

  const int widx = blockIdx.x * 8 + wv;        // 0 .. B*H*(T/16)-1
  const int qb   = widx & (T_SEQ / 16 - 1);    // query chunk in sequence
  const int bh   = widx >> 7;                  // fused (batch,head): T/16=128
  const int q0   = qb * 16;

  const _Float16* qbase = qh + (size_t)bh * T_SEQ * D_HEAD;
  const _Float16* kbase = kh + (size_t)bh * T_SEQ * D_HEAD;
  const _Float16* vbase = vT + (size_t)bh * D_HEAD * T_SEQ;

  // Persistent q A-fragments for K-dim chunks {0..31}, {32..63}
  const _Float16* qrow = qbase + (size_t)(q0 + l15) * D_HEAD;
  v16h qf0 = ld_frag(qrow + 8 * hl,      qrow + 16 + 8 * hl);
  v16h qf1 = ld_frag(qrow + 32 + 8 * hl, qrow + 48 + 8 * hl);

  const float NEG_INF = -__builtin_inff();
  float m_r[8], l_r[8];
  v8f z = {};
  v8f accd[4];
#pragma unroll
  for (int dt = 0; dt < 4; ++dt) accd[dt] = z;
#pragma unroll
  for (int r = 0; r < 8; ++r) { m_r[r] = NEG_INF; l_r[r] = 0.0f; }

  __shared__ __align__(16) _Float16 Pl[8][16 * 32];  // 1KB per wave
  _Float16* pw = &Pl[wv][0];

  for (int k0 = 0; k0 <= q0 + 15; k0 += 32) {
    // ---- S = q @ k^T  (two 16x16 N-tiles) ----
    v8f s[2];
#pragma unroll
    for (int nt = 0; nt < 2; ++nt) {
      const _Float16* krow = kbase + (size_t)(k0 + nt * 16 + l15) * D_HEAD;
      v16h kf0 = ld_frag(krow + 8 * hl,      krow + 16 + 8 * hl);
      v16h kf1 = ld_frag(krow + 32 + 8 * hl, krow + 48 + 8 * hl);
      v8f c = z;
      c = WMMA16(qf0, kf0, c);
      c = WMMA16(qf1, kf1, c);
      s[nt] = c;
    }
    // ---- causal mask (diagonal blocks only) ----
    if (k0 + 31 > q0) {
#pragma unroll
      for (int nt = 0; nt < 2; ++nt)
#pragma unroll
        for (int r = 0; r < 8; ++r) {
          int key = k0 + nt * 16 + l15;
          int qry = q0 + r + 8 * hl;
          s[nt][r] = (key > qry) ? NEG_INF : s[nt][r];
        }
    }
    // ---- online softmax: row max / exp / row sum (rows live across 16 lanes) ----
    float mnew[8], rs[8];
#pragma unroll
    for (int r = 0; r < 8; ++r) {
      float mx = fmaxf(s[0][r], s[1][r]);
#pragma unroll
      for (int d = 1; d < 16; d <<= 1) mx = fmaxf(mx, __shfl_xor(mx, d, 32));
      mnew[r] = fmaxf(m_r[r], mx);
      s[0][r] = __expf(s[0][r] - mnew[r]);
      s[1][r] = __expf(s[1][r] - mnew[r]);
      float sum = s[0][r] + s[1][r];
#pragma unroll
      for (int d = 1; d < 16; d <<= 1) sum += __shfl_xor(sum, d, 32);
      rs[r] = sum;
    }
#pragma unroll
    for (int r = 0; r < 8; ++r) {
      float alpha = __expf(m_r[r] - mnew[r]);
      l_r[r] = l_r[r] * alpha + rs[r];
      m_r[r] = mnew[r];
#pragma unroll
      for (int dt = 0; dt < 4; ++dt) accd[dt][r] *= alpha;
    }
    // ---- re-layout P: C-tile layout -> A-fragment layout via wave-private LDS ----
#pragma unroll
    for (int nt = 0; nt < 2; ++nt)
#pragma unroll
      for (int r = 0; r < 8; ++r)
        pw[(r + 8 * hl) * 32 + nt * 16 + l15] = (_Float16)s[nt][r];
    asm volatile("s_wait_dscnt 0" ::: "memory");
    const _Float16* prow = pw + l15 * 32;
    v16h pf = ld_frag(prow + 8 * hl, prow + 16 + 8 * hl);
    // ---- acc += P @ V  (V^T rows are contiguous along keys) ----
#pragma unroll
    for (int dt = 0; dt < 4; ++dt) {
      const _Float16* vrow = vbase + (size_t)(dt * 16 + l15) * T_SEQ + k0 + 8 * hl;
      v16h vf = ld_frag(vrow, vrow + 16);
      accd[dt] = WMMA16(pf, vf, accd[dt]);
    }
  }

  // ---- normalize and emit y in [B,T,C] f16 for the projection GEMM ----
  const int bb = bh >> 4, hh = bh & 15;
#pragma unroll
  for (int r = 0; r < 8; ++r) {
    float inv = 1.0f / l_r[r];
    int t = q0 + r + 8 * hl;
#pragma unroll
    for (int dt = 0; dt < 4; ++dt)
      yh[((size_t)(bb * T_SEQ + t)) * C_DIM + hh * D_HEAD + dt * 16 + l15] =
          (_Float16)(accd[dt][r] * inv);
  }
}

// ---------------------------------------------------------------------------
// Kernel 4: projection GEMM. A=yh [M_TOK,C] f16, Bt=wprojT [C,C] f16, f32 out.
__global__ void gemm_proj_kernel(const _Float16* __restrict__ A,
                                 const _Float16* __restrict__ Bt,
                                 const float* __restrict__ bias,
                                 float* __restrict__ out) {
  const int lane = threadIdx.x & 31;
  const int wv   = threadIdx.x >> 5;
  const int hl   = lane >> 4;
  const int l15  = lane & 15;
  const int n0   = blockIdx.x * 64;
  const int m0   = blockIdx.y * 128 + wv * 32;

  const _Float16* arow0 = A + (size_t)(m0 + l15) * C_DIM;
  const _Float16* arow1 = A + (size_t)(m0 + 16 + l15) * C_DIM;
  const _Float16* bcol[4];
#pragma unroll
  for (int nt = 0; nt < 4; ++nt)
    bcol[nt] = Bt + (size_t)(n0 + nt * 16 + l15) * C_DIM;

  v8f z = {};
  v8f acc[2][4];
#pragma unroll
  for (int i = 0; i < 2; ++i)
#pragma unroll
    for (int nt = 0; nt < 4; ++nt) acc[i][nt] = z;

  for (int kb = 0; kb < C_DIM; kb += 32) {
    __builtin_prefetch(arow0 + kb + 512, 0, 1);
    v16h a0 = ld_frag(arow0 + kb + 8 * hl, arow0 + kb + 16 + 8 * hl);
    v16h a1 = ld_frag(arow1 + kb + 8 * hl, arow1 + kb + 16 + 8 * hl);
#pragma unroll
    for (int nt = 0; nt < 4; ++nt) {
      const _Float16* bp = bcol[nt] + kb + 8 * hl;
      v16h b = ld_frag(bp, bp + 16);
      acc[0][nt] = WMMA16(a0, b, acc[0][nt]);
      acc[1][nt] = WMMA16(a1, b, acc[1][nt]);
    }
  }

  float bs[4];
#pragma unroll
  for (int nt = 0; nt < 4; ++nt) bs[nt] = bias[n0 + nt * 16 + l15];
#pragma unroll
  for (int i = 0; i < 2; ++i)
#pragma unroll
    for (int nt = 0; nt < 4; ++nt)
#pragma unroll
      for (int r = 0; r < 8; ++r) {
        int row = m0 + i * 16 + r + 8 * hl;
        out[(size_t)row * C_DIM + n0 + nt * 16 + l15] = acc[i][nt][r] + bs[nt];
      }
}

// ---------------------------------------------------------------------------
extern "C" void kernel_launch(void* const* d_in, const int* in_sizes, int n_in,
                              void* d_out, int out_size, void* d_ws, size_t ws_size,
                              hipStream_t stream) {
  (void)in_sizes; (void)n_in; (void)out_size; (void)ws_size;
  const float* x      = (const float*)d_in[0];   // [B,T,C]
  const float* w_qkv  = (const float*)d_in[1];   // [C,3C]
  const float* b_qkv  = (const float*)d_in[2];   // [3C]
  const float* w_proj = (const float*)d_in[3];   // [C,C]
  const float* b_proj = (const float*)d_in[4];   // [C]
  float* out = (float*)d_out;                    // [B,T,C]

  char* ws = (char*)d_ws;
  _Float16* xh     = (_Float16*)(ws + (size_t)( 0u << 20));  // 8 MB  [M,C]
  _Float16* wqkvT  = (_Float16*)(ws + (size_t)( 8u << 20));  // 6 MB  [3C,C]
  _Float16* wprojT = (_Float16*)(ws + (size_t)(14u << 20));  // 2 MB  [C,C]
  _Float16* qh     = (_Float16*)(ws + (size_t)(16u << 20));  // 8 MB  [B,H,T,D]
  _Float16* kh     = (_Float16*)(ws + (size_t)(24u << 20));  // 8 MB  [B,H,T,D]
  _Float16* vT     = (_Float16*)(ws + (size_t)(32u << 20));  // 8 MB  [B,H,D,T]
  _Float16* yh     = (_Float16*)(ws + (size_t)(40u << 20));  // 8 MB  [M,C]

  const int nx = M_TOK * C_DIM;
  cvt_f16_kernel<<<(nx + 255) / 256, 256, 0, stream>>>(x, xh, nx);
  transpose_f16_kernel<<<(C_DIM * N_QKV + 255) / 256, 256, 0, stream>>>(
      w_qkv, wqkvT, C_DIM, N_QKV);
  transpose_f16_kernel<<<(C_DIM * C_DIM + 255) / 256, 256, 0, stream>>>(
      w_proj, wprojT, C_DIM, C_DIM);

  gemm_qkv_kernel<<<dim3(N_QKV / 64, M_TOK / 128), 128, 0, stream>>>(
      xh, wqkvT, b_qkv, qh, kh, vT);

  flash_attn_kernel<<<dim3(BATCH * H_NUM * (T_SEQ / 16) / 8), 256, 0, stream>>>(
      qh, kh, vT, yh);

  gemm_proj_kernel<<<dim3(C_DIM / 64, M_TOK / 128), 128, 0, stream>>>(
      yh, wprojT, b_proj, out);
}
